// CircularLinear_31834297598770
// MI455X (gfx1250) — matrix operand order repaired and checked by reading, not compiled, not run on hardware
//
#include <hip/hip_runtime.h>
#include <hip/hip_bf16.h>

// ---------------------------------------------------------------------------
// CircularLinear as circulant GEMM:
//   Y[m,n] = sum_c X[m,c] * W[(n-c) mod 2048],  W = Wf*Ws
//   M = 8*64*64 = 32768, K = 2048, N = 1536
// Kernel 1: build Bt[n][c] = bf16(W[(n-c)&2047])  (6 MB, L2-resident, in d_ws)
// Kernel 2: WMMA bf16 GEMM, 128x128 block tile, K-step 32, double-buffered
//           LDS fed by gfx1250 async global->LDS copies.
// ---------------------------------------------------------------------------

#define IN_DIM   2048
#define OUT_DIM  1536
#define M_ROWS   32768

typedef __attribute__((ext_vector_type(16))) __bf16 v16bf;
typedef __attribute__((ext_vector_type(8)))  float  v8f;
typedef __attribute__((ext_vector_type(4)))  int    v4i;

#define AS1 __attribute__((address_space(1)))
#define AS3 __attribute__((address_space(3)))

#if defined(__has_builtin)
#  if __has_builtin(__builtin_amdgcn_global_load_async_to_lds_b128)
#    define HAS_ASYNC_CP 1
#  endif
#endif
#ifndef HAS_ASYNC_CP
#  define HAS_ASYNC_CP 0
#endif

__device__ __forceinline__ void wait_async0() {
#if defined(__has_builtin) && __has_builtin(__builtin_amdgcn_s_wait_asynccnt)
  __builtin_amdgcn_s_wait_asynccnt(0);
#elif defined(__AMDGCN__)
  asm volatile("s_wait_asynccnt 0" ::: "memory");
#endif
}

#if HAS_ASYNC_CP
__device__ __forceinline__ void async_cp_b128(const void* g, void* l) {
  __builtin_amdgcn_global_load_async_to_lds_b128(
      (AS1 v4i*)(AS1 void*)g, (AS3 v4i*)(AS3 void*)l, 0, 0);
}
#endif

// LDS row strides (padded to kill bank conflicts on fragment reads)
#define LDA 36   // floats per A row (32 data + 4 pad)  -> 144 B
#define LDB 40   // bf16  per B row (32 data + 8 pad)  ->  80 B

// ---------------------------------------------------------------------------
// Kernel 1: build circulant matrix Bt[n][c] = bf16(Wf[k]*Ws[k]), k=(n-c)&2047
// ---------------------------------------------------------------------------
__global__ __launch_bounds__(256) void cl_build_bt(const float* __restrict__ wf,
                                                   const float* __restrict__ ws,
                                                   __bf16* __restrict__ bt) {
  int i = blockIdx.x * 256 + threadIdx.x;      // 1536*512 threads, 4 elems each
  int n  = i >> 9;
  int c0 = (i & 511) << 2;
  __attribute__((aligned(8))) __bf16 v[4];
#pragma unroll
  for (int j = 0; j < 4; ++j) {
    int k = (n - (c0 + j)) & (IN_DIM - 1);
    v[j] = (__bf16)(wf[k] * ws[k]);
  }
  *(uint2*)(&bt[(size_t)n * IN_DIM + c0]) = *(const uint2*)v;
}

// ---------------------------------------------------------------------------
// Kernel 2: GEMM  Y[M, N] = X[M, K] * Bt^T   (Bt stored [N][K])
// ---------------------------------------------------------------------------
__global__ __launch_bounds__(256) void cl_wmma_gemm(const float* __restrict__ X,
                                                    const __bf16* __restrict__ Bt,
                                                    float* __restrict__ Y) {
  __shared__ float  sA[2][128 * LDA];   // 36 KB
  __shared__ __bf16 sB[2][128 * LDB];   // 20 KB

  const int tid  = threadIdx.x;
  const int lane = tid & 31;
  const int wave = tid >> 5;
  const int w_m  = wave & 3;            // 4 wave-rows  * 32 rows
  const int w_n  = wave >> 2;           // 2 wave-cols  * 64 cols
  const int half = lane >> 4;           // K-phase selector per WMMA layout
  const int lr   = lane & 15;

  const int blockN = blockIdx.x * 128;  // x fastest -> 12 N-blocks share A in L2
  const int blockM = blockIdx.y * 128;

  v8f acc[2][4];
#pragma unroll
  for (int mt = 0; mt < 2; ++mt)
#pragma unroll
    for (int nt = 0; nt < 4; ++nt)
#pragma unroll
      for (int r = 0; r < 8; ++r) acc[mt][nt][r] = 0.0f;

  const int KSTEPS = IN_DIM / 32;       // 64

#if HAS_ASYNC_CP
  auto prefetch = [&](int s, int b) {
    const int k0 = s * 32;
    // A tile: 128 rows x 32 f32 = 1024 x 16B chunks, 4 per thread
#pragma unroll
    for (int i = 0; i < 4; ++i) {
      int c = tid + i * 256;
      int row = c >> 3, off = c & 7;
      const float* g = X + (size_t)(blockM + row) * IN_DIM + k0 + off * 4;
      float* l = &sA[b][row * LDA + off * 4];
      async_cp_b128(g, l);
    }
    // B tile: 128 rows x 32 bf16 = 512 x 16B chunks, 2 per thread
#pragma unroll
    for (int i = 0; i < 2; ++i) {
      int c = tid + i * 256;
      int row = c >> 2, off = c & 3;
      const __bf16* g = Bt + (size_t)(blockN + row) * IN_DIM + k0 + off * 8;
      __bf16* l = &sB[b][row * LDB + off * 8];
      async_cp_b128(g, l);
    }
  };
  prefetch(0, 0);
#endif

  for (int s = 0; s < KSTEPS; ++s) {
#if HAS_ASYNC_CP
    const int buf = s & 1;
    wait_async0();                 // my copies for buf are done
    __syncthreads();               // everyone's copies done / prev compute done
    if (s + 1 < KSTEPS) prefetch(s + 1, buf ^ 1);   // overlap with compute
#else
    const int buf = 0;
    __syncthreads();
#pragma unroll
    for (int i = 0; i < 4; ++i) {
      int c = tid + i * 256;
      int row = c >> 3, off = c & 7;
      *(float4*)&sA[0][row * LDA + off * 4] =
          *(const float4*)(X + (size_t)(blockM + row) * IN_DIM + s * 32 + off * 4);
    }
#pragma unroll
    for (int i = 0; i < 2; ++i) {
      int c = tid + i * 256;
      int row = c >> 2, off = c & 3;
      *(float4*)&sB[0][row * LDB + off * 8] =
          *(const float4*)(Bt + (size_t)(blockN + row) * IN_DIM + s * 32 + off * 8);
    }
    __syncthreads();
#endif

    // ---- A fragments (16x32 bf16): lane half selects K {0..7,16..23} vs {8..15,24..31}
    v16bf afr[2];
    const int klo = half ? 8 : 0;
#pragma unroll
    for (int mt = 0; mt < 2; ++mt) {
      const float* base = &sA[buf][(w_m * 32 + mt * 16 + lr) * LDA + klo];
      float f[16];
      *(float4*)&f[0]  = *(const float4*)&base[0];
      *(float4*)&f[4]  = *(const float4*)&base[4];
      *(float4*)&f[8]  = *(const float4*)&base[16];
      *(float4*)&f[12] = *(const float4*)&base[20];
      v16bf a;
#pragma unroll
      for (int i = 0; i < 16; ++i) a[i] = (__bf16)f[i];
      afr[mt] = a;
    }

    // ---- B fragments (32x16 bf16): lane half selects K {0..7,16..23} vs {8..15,24..31}
    v16bf bfr[4];
    const int kb = half ? 8 : 0;
#pragma unroll
    for (int nt = 0; nt < 4; ++nt) {
      const __bf16* bb = &sB[buf][(w_n * 64 + nt * 16 + lr) * LDB + kb];
      union { float4 q[2]; v16bf v; } u;
      u.q[0] = *(const float4*)&bb[0];
      u.q[1] = *(const float4*)&bb[16];
      bfr[nt] = u.v;
    }

    // ---- 8 WMMAs per wave per K-step
#pragma unroll
    for (int mt = 0; mt < 2; ++mt)
#pragma unroll
      for (int nt = 0; nt < 4; ++nt)
        acc[mt][nt] = __builtin_amdgcn_wmma_f32_16x16x32_bf16(
            false, afr[mt], false, bfr[nt], (short)0, acc[mt][nt], false, false);
  }

  // ---- Epilogue: C/D layout -> lane half selects M-row 0..7 vs 8..15
#pragma unroll
  for (int mt = 0; mt < 2; ++mt)
#pragma unroll
    for (int nt = 0; nt < 4; ++nt) {
      const int m0 = blockM + w_m * 32 + mt * 16 + half * 8;
      const int n0 = blockN + w_n * 64 + nt * 16 + lr;
      float* yp = Y + (size_t)m0 * OUT_DIM + n0;
#pragma unroll
      for (int r = 0; r < 8; ++r) yp[(size_t)r * OUT_DIM] = acc[mt][nt][r];
    }
}

// ---------------------------------------------------------------------------
extern "C" void kernel_launch(void* const* d_in, const int* in_sizes, int n_in,
                              void* d_out, int out_size, void* d_ws, size_t ws_size,
                              hipStream_t stream) {
  const float* x  = (const float*)d_in[0];
  const float* wf = (const float*)d_in[1];
  const float* ws = (const float*)d_in[2];
  float* out  = (float*)d_out;
  __bf16* bt  = (__bf16*)d_ws;          // 1536*2048*2 = 6 MB

  cl_build_bt<<<(OUT_DIM * (IN_DIM / 4)) / 256, 256, 0, stream>>>(wf, ws, bt);

  dim3 grid(OUT_DIM / 128, M_ROWS / 128);   // (12, 256), x fastest for L2 reuse
  cl_wmma_gemm<<<grid, 256, 0, stream>>>(x, bt, out);
}
